// DynamicDualStreamAttention_39161511804961
// MI455X (gfx1250) — compile-verified, matrix operand order checked by reading, not run on hardware
//
#include <hip/hip_runtime.h>
#include <hip/hip_bf16.h>
#include <math.h>
#include <stdint.h>

#define DEV static __device__ __forceinline__

typedef __attribute__((ext_vector_type(16))) __bf16 bf16x16;
typedef __attribute__((ext_vector_type(8)))  float  f32x8;
typedef __attribute__((ext_vector_type(4)))  unsigned int u32x4;
typedef __attribute__((ext_vector_type(4)))  int i32x4;
typedef __attribute__((ext_vector_type(8)))  int i32x8;

// ---- problem constants ----
static constexpr int  NB   = 64;     // batch
static constexpr int  NC   = 256;    // channels
static constexpr int  NT   = 64;     // time
static constexpr int  NV   = 25;     // vertices
static constexpr int  NH   = 4;      // heads
static constexpr int  NR   = 32;     // rank
static constexpr int  NS   = 3;      // subsets
static constexpr int  NGH  = 64;     // gate hidden
static constexpr int  NHOP = 13;     // MAXHOP+1
static constexpr int  TV   = NT * NV;             // 1600
static constexpr long NEX  = (long)NB * NC * TV;  // 26,214,400
static constexpr int  MR   = NB * NT * NV;        // 102400 global-stream rows
static constexpr long ABFN = (long)NB * NC * NV * NV;  // 10,240,000
static constexpr long X4N  = (long)NB * NT * NV * NV;  //  2,560,000

// ---- bf16 helpers (storage = unsigned short) ----
DEV unsigned short f2bf(float f) {
  unsigned int u = __float_as_uint(f);
  unsigned int r = u + 0x7fffu + ((u >> 16) & 1u);
  return (unsigned short)(r >> 16);
}
DEV float bf2f(unsigned short h) { return __uint_as_float(((unsigned int)h) << 16); }

DEV unsigned lds_off(const void* p) { return (unsigned)(uintptr_t)p; }

// =====================================================================
// Tensor Data Mover: 2D tile load global->LDS (D# per CDNA5 ISA)
// OOB (x >= ten_w or y >= ten_h) reads return zero -> free K/N padding.
// data_size code 1 = 2-byte elements (bf16). All dims in elements.
// =====================================================================
DEV void tdm_load_2d(unsigned ldsaddr, const void* gptr,
                     unsigned ten_w, unsigned ten_h,
                     unsigned tile_w, unsigned tile_h, unsigned stride) {
  asm volatile("" ::: "memory");
  unsigned long long ga = (unsigned long long)(uintptr_t)gptr;
  u32x4 g0;
  g0[0] = 1u;                                    // count=1, is_restore=0, gather=0
  g0[1] = ldsaddr;                               // lds_addr [63:32]
  g0[2] = (unsigned)ga;                          // global_addr lo
  g0[3] = (unsigned)((ga >> 32) & 0x1ffffffu) | (2u << 30);  // addr hi | type=2
  i32x8 g1;
  g1[0] = (int)(1u << 16);                       // data_size=1 (2B)
  g1[1] = (int)((ten_w & 0xffffu) << 16);        // tensor_dim0 lo16 @ bits 63:48
  g1[2] = (int)((ten_w >> 16) | ((ten_h & 0xffffu) << 16));   // dim0 hi | dim1 lo
  g1[3] = (int)((ten_h >> 16) | ((tile_w & 0xffffu) << 16));  // dim1 hi | tile_dim0
  g1[4] = (int)(tile_h & 0xffffu);               // tile_dim1 (tile_dim2=0)
  g1[5] = (int)stride;                           // tensor_dim0_stride lo32
  g1[6] = 0;
  g1[7] = 0;
  i32x4 z4 = {0, 0, 0, 0};
#if __clang_major__ >= 23
  i32x8 z8 = {0, 0, 0, 0, 0, 0, 0, 0};
  __builtin_amdgcn_tensor_load_to_lds(g0, g1, z4, z4, z8, 0);
#else
  __builtin_amdgcn_tensor_load_to_lds(g0, g1, z4, z4, 0);
#endif
}
DEV void tdm_wait0() { __builtin_amdgcn_s_wait_tensorcnt(0); asm volatile("" ::: "memory"); }
DEV void tdm_wait2() { __builtin_amdgcn_s_wait_tensorcnt(2); asm volatile("" ::: "memory"); }

// ---- WMMA fragment loader (16x32 bf16 tile from LDS, row-major) ----
//  lanes 0-15: row=lane,    K = {0..7, 16..23}
//  lanes16-31: row=lane-16, K = {8..15, 24..31}
DEV bf16x16 frag_ld(const unsigned short* sm, int ld, int lane) {
  const int r  = lane & 15;
  const int pb = (lane & 16) ? 4 : 0;
  const unsigned int* row = (const unsigned int*)(sm + r * ld);
  union { unsigned int u[8]; bf16x16 v; } f;
#pragma unroll
  for (int j = 0; j < 4; ++j) { f.u[j] = row[pb + j]; f.u[4 + j] = row[8 + pb + j]; }
  return f.v;
}

DEV f32x8 wmma32(bf16x16 a, bf16x16 b, f32x8 c) {
  return __builtin_amdgcn_wmma_f32_16x16x32_bf16(false, a, false, b, (short)0, c, false, false);
}

// =====================================================================
// Generic NT GEMM with double-buffered TDM staging.
// out(m,n) = sum_k A[m,k]*B[n,k]; A (M,K), B (N,K) bf16 row-major.
// Block tile 64x64, 8 waves, K step 32, grid=(M/64, N/64, batch).
// =====================================================================
template <typename Epi>
__global__ __launch_bounds__(256) void gemm_nt(const unsigned short* __restrict__ A,
                                               const unsigned short* __restrict__ Bw,
                                               int K, long aBatch, Epi epi) {
  __shared__ __align__(16) unsigned short sA[2][64 * 32];
  __shared__ __align__(16) unsigned short sB[2][64 * 32];
  const int bz = blockIdx.z;
  const unsigned short* Ab = A + (long)bz * aBatch;
  const int m0 = blockIdx.x * 64, n0 = blockIdx.y * 64;
  const int tid = threadIdx.x, lane = tid & 31, wv = tid >> 5;
  const int wr = wv >> 1, wc = wv & 1;
  const bool w0 = (__builtin_amdgcn_readfirstlane(wv) == 0);
  f32x8 acc0 = {0, 0, 0, 0, 0, 0, 0, 0}, acc1 = acc0;
  const int nk = K >> 5;
  if (w0) {
    tdm_load_2d(lds_off(&sA[0][0]), Ab + (long)m0 * K, 32, 64, 32, 64, K);
    tdm_load_2d(lds_off(&sB[0][0]), Bw + (long)n0 * K, 32, 64, 32, 64, K);
  }
  for (int ks = 0; ks < nk; ++ks) {
    const int cur = ks & 1;
    if (w0) {
      if (ks + 1 < nk) {
        const int nxt = (ks + 1) & 1;
        tdm_load_2d(lds_off(&sA[nxt][0]), Ab + (long)m0 * K + (ks + 1) * 32, 32, 64, 32, 64, K);
        tdm_load_2d(lds_off(&sB[nxt][0]), Bw + (long)n0 * K + (ks + 1) * 32, 32, 64, 32, 64, K);
        tdm_wait2();   // two just-issued outstanding; tiles for ks are done
      } else {
        tdm_wait0();
      }
    }
    __syncthreads();
    bf16x16 af = frag_ld(sA[cur] + wr * 16 * 32, 32, lane);
    bf16x16 b0 = frag_ld(sB[cur] + (wc * 32) * 32, 32, lane);
    bf16x16 b1 = frag_ld(sB[cur] + (wc * 32 + 16) * 32, 32, lane);
    acc0 = wmma32(af, b0, acc0);
    acc1 = wmma32(af, b1, acc1);
    __syncthreads();   // readers done before next TDM overwrites the buffer
  }
  const int mb = m0 + wr * 16 + 8 * (lane >> 4);
  const int nb = n0 + wc * 32 + (lane & 15);
#pragma unroll
  for (int r = 0; r < 8; ++r) {
    epi(bz, mb + r, nb, acc0[r]);
    epi(bz, mb + r, nb + 16, acc1[r]);
  }
}

// ---- epilogues ----
struct EpiX3 {  // x3bf[b, n(=o), m(=tv)] = bf16(acc + c3b[n])   (transposed store)
  const float* bias; unsigned short* out;
  __device__ void operator()(int b, int m, int n, float acc) const {
    out[(long)b * (NC * TV) + (long)n * TV + m] = f2bf(acc + bias[n]);
  }
};
struct EpiQKV {  // split 768 cols into q/k/v bf16 (row-major, 256 each)
  unsigned short *q, *k, *v;
  __device__ void operator()(int, int m, int n, float acc) const {
    unsigned short hv = f2bf(acc);
    if (n < 256)      q[(long)m * 256 + n] = hv;
    else if (n < 512) k[(long)m * 256 + (n - 256)] = hv;
    else              v[(long)m * 256 + (n - 512)] = hv;
  }
};
struct EpiOutProj {  // g[m,n] = acc + out_b[n] + xg[m,n]
  const float* bias; const float* x; float* g;
  __device__ void operator()(int, int m, int n, float acc) const {
    int b = m / TV, tv = m % TV;
    g[(long)m * NC + n] = acc + bias[n] + x[((long)b * NC + n) * TV + tv];
  }
};
struct EpiG2Fuse {  // gate=sigmoid(acc+b); fused = gate*lp + (1-gate)*g4
  const float* bias; const float* localp; const float* g; float* fused;
  __device__ void operator()(int, int m, int n, float acc) const {
    float gt = 1.f / (1.f + expf(-(acc + bias[n])));
    int b = m / TV, tv = m % TV;
    float lp = localp[((long)b * NC + n) * TV + tv];
    float g4 = g[(long)m * NC + n];
    fused[((long)b * NC + n) * TV + tv] = gt * lp + (1.f - gt) * g4;
  }
};

// =====================================================================
// g1 GEMM with fused concat: A[m, k<256]=local (gathered), A[m, k>=256]=g.
// B = g1w (64 x 512) bf16 via TDM. N = 64 (one tile). hg = relu(acc+bias).
// =====================================================================
__global__ __launch_bounds__(256) void gemm_g1(const float* __restrict__ localp,
                                               const float* __restrict__ g,
                                               const unsigned short* __restrict__ Bw,
                                               const float* __restrict__ bias,
                                               float* __restrict__ hg) {
  __shared__ __align__(16) unsigned short sA[64 * 32];
  __shared__ __align__(16) unsigned short sB[64 * 32];
  const int m0 = blockIdx.x * 64;
  const int tid = threadIdx.x, lane = tid & 31, wv = tid >> 5;
  const int wr = wv >> 1, wc = wv & 1;
  const bool w0 = (__builtin_amdgcn_readfirstlane(wv) == 0);
  f32x8 acc0 = {0, 0, 0, 0, 0, 0, 0, 0}, acc1 = acc0;
  const int m = tid & 63, cg = tid >> 6;  // 64 rows x 4 col-groups of 8
  const int gm = m0 + m;
  const int b = gm / TV, tv = gm % TV;
  for (int k0 = 0; k0 < 2 * NC; k0 += 32) {
    if (w0) tdm_load_2d(lds_off(&sB[0]), Bw + k0, 32, 64, 32, 64, 2 * NC);
    if (k0 < NC) {  // local stream, channel-major gather (coalesced along m)
      const float* lp = localp + ((long)b * NC + k0 + cg * 8) * TV + tv;
#pragma unroll
      for (int j = 0; j < 8; ++j) sA[m * 32 + cg * 8 + j] = f2bf(lp[(long)j * TV]);
    } else {        // global stream, row-major
      const float* gp = g + (long)gm * NC + (k0 - NC) + cg * 8;
#pragma unroll
      for (int j = 0; j < 8; ++j) sA[m * 32 + cg * 8 + j] = f2bf(gp[j]);
    }
    if (w0) tdm_wait0();
    __syncthreads();
    bf16x16 af = frag_ld(sA + wr * 16 * 32, 32, lane);
    bf16x16 b0 = frag_ld(sB + (wc * 32) * 32, 32, lane);
    bf16x16 b1 = frag_ld(sB + (wc * 32 + 16) * 32, 32, lane);
    acc0 = wmma32(af, b0, acc0);
    acc1 = wmma32(af, b1, acc1);
    __syncthreads();
  }
  const int mb = m0 + wr * 16 + 8 * (lane >> 4);
  const int nb = wc * 32 + (lane & 15);
#pragma unroll
  for (int r = 0; r < 8; ++r) {
    hg[(long)(mb + r) * NGH + nb]      = fmaxf(acc0[r] + bias[nb], 0.f);
    hg[(long)(mb + r) * NGH + nb + 16] = fmaxf(acc1[r] + bias[nb + 16], 0.f);
  }
}

// =====================================================================
// Small prep kernels
// =====================================================================
__global__ __launch_bounds__(256) void cvt_f32_bf16(const float* __restrict__ s,
                                                    unsigned short* __restrict__ d, long n) {
  long i = (long)blockIdx.x * 256 + threadIdx.x;
  if (i < n) d[i] = f2bf(s[i]);
}

__global__ __launch_bounds__(256) void xt_kernel(const float* __restrict__ x,
                                                 unsigned short* __restrict__ xt) {
  long i = (long)blockIdx.x * 256 + threadIdx.x;
  long m = i >> 8; int c = (int)(i & 255);
  int b = (int)(m / TV), tv = (int)(m % TV);
  xt[i] = f2bf(x[((long)b * NC + c) * TV + tv]);
}

__global__ __launch_bounds__(256) void xmean_kernel(const float* __restrict__ x,
                                                    float* __restrict__ xm) {
  int i = blockIdx.x * 256 + threadIdx.x;
  int b = i / (NC * NV), rem = i % (NC * NV);
  int c = rem / NV, v = rem % NV;
  const float* p = x + ((long)b * NC + c) * TV + v;
  float s = 0.f;
  for (int t = 0; t < NT; ++t) s += p[t * NV];
  xm[i] = s * (1.f / NT);
}

__global__ __launch_bounds__(256) void w3m_kernel(const float* __restrict__ c3w,
                                                  const float* __restrict__ c3b,
                                                  float* __restrict__ w3m,
                                                  float* __restrict__ b3m) {
  int idx = blockIdx.x * 256 + threadIdx.x;
  if (idx >= NS * (NC + 1)) return;
  int i = idx / (NC + 1), j = idx % (NC + 1);
  if (j < NC) {
    float s = 0.f;
    for (int o = 0; o < NC; ++o) s += c3w[((long)i * NC + o) * NC + j];
    w3m[i * NC + j] = s * (1.f / NC);
  } else {
    float s = 0.f;
    for (int o = 0; o < NC; ++o) s += c3b[i * NC + o];
    b3m[i] = s * (1.f / NC);
  }
}

__global__ __launch_bounds__(256) void x1m_kernel(const float* __restrict__ c1w,
                                                  const float* __restrict__ c1b,
                                                  const float* __restrict__ xm,
                                                  float* __restrict__ x1m, int i) {
  int idx = blockIdx.x * 256 + threadIdx.x;
  int b = idx / (NR * NV), rem = idx % (NR * NV);
  int r = rem / NV, v = rem % NV;
  const float* w = c1w + ((long)i * NR + r) * NC;
  const float* xb = xm + (long)b * NC * NV + v;
  float s = c1b[i * NR + r];
  for (int c = 0; c < NC; ++c) s += w[c] * xb[c * NV];
  x1m[idx] = s;
}

__global__ __launch_bounds__(256) void a_kernel(const float* __restrict__ c4w,
                                                const float* __restrict__ c4b,
                                                const float* __restrict__ x1m,
                                                const float* __restrict__ PA,
                                                const float* __restrict__ alphaP,
                                                unsigned short* __restrict__ ab, int i) {
  __shared__ float dsh[NR];
  int blk = blockIdx.x;
  int b = blk / (NV * NV), uv = blk % (NV * NV);
  int u = uv / NV, v = uv % NV;
  int o = threadIdx.x;
  if (o < NR) {
    const float* xb = x1m + (long)b * NR * NV;
    dsh[o] = tanhf(xb[o * NV + u] - xb[o * NV + v]);
  }
  __syncthreads();
  const float* w = c4w + ((long)i * NC + o) * NR;
  float s = c4b[i * NC + o];
#pragma unroll
  for (int r = 0; r < NR; ++r) s += w[r] * dsh[r];
  float a = s * alphaP[0] + PA[((long)i * NV + u) * NV + v];
  ab[((long)b * NC + o) * (NV * NV) + u * NV + v] = f2bf(a);
}

__global__ __launch_bounds__(256) void cm_kernel(const float* __restrict__ x,
                                                 const float* __restrict__ w3m,
                                                 const float* __restrict__ b3m,
                                                 float* __restrict__ cm, int i) {
  int idx = blockIdx.x * 256 + threadIdx.x;
  int b = idx / TV, tv = idx % TV;
  const float* w = w3m + i * NC;
  const float* xb = x + (long)b * NC * TV + tv;
  float s = b3m[i];
  for (int c = 0; c < NC; ++c) s += w[c] * xb[(long)c * TV];
  cm[idx] = s;
}

__global__ __launch_bounds__(256) void x4_kernel(const float* __restrict__ cm,
                                                 unsigned short* __restrict__ x4b) {
  long idx = (long)blockIdx.x * 256 + threadIdx.x;
  long bt = idx / (NV * NV); int mn = (int)(idx % (NV * NV));
  int m = mn / NV, n = mn % NV;
  x4b[idx] = f2bf(tanhf(cm[bt * NV + m] - cm[bt * NV + n]));
}

// =====================================================================
// y1 (all subsets, register accumulation): per (b,c)
//   y[t,u] = beta * sum_i sum_v x3_i[t,v]*a_i[u,v]
// TDM per-wave tile loads; OOB-zero gives the 25->32 padding.
// =====================================================================
__global__ __launch_bounds__(256) void y1_kernel(const unsigned short* __restrict__ x3b,
                                                 const unsigned short* __restrict__ ab,
                                                 float* __restrict__ y,
                                                 const float* __restrict__ betaP) {
  __shared__ __align__(16) unsigned short sA[8][2048];  // 64 x 32
  __shared__ __align__(16) unsigned short sB[8][1024];  // 32 x 32
  const int wv = threadIdx.x >> 5, lane = threadIdx.x & 31;
  const long bc = (long)blockIdx.x * 8 + wv;
  const unsigned aoff = lds_off(&sA[wv][0]);
  const unsigned boff = lds_off(&sB[wv][0]);
  f32x8 acc[4][2];
#pragma unroll
  for (int mt = 0; mt < 4; ++mt)
#pragma unroll
    for (int nt = 0; nt < 2; ++nt) acc[mt][nt] = (f32x8){0, 0, 0, 0, 0, 0, 0, 0};
  for (int i = 0; i < NS; ++i) {
    tdm_load_2d(aoff, x3b + (long)i * NEX + bc * TV, 25, 64, 32, 64, 25);
    tdm_load_2d(boff, ab + (long)i * ABFN + bc * (NV * NV), 25, 25, 32, 32, 25);
    tdm_wait0();
#pragma unroll
    for (int mt = 0; mt < 4; ++mt) {
      bf16x16 af = frag_ld(sA[wv] + mt * 16 * 32, 32, lane);
#pragma unroll
      for (int nt = 0; nt < 2; ++nt) {
        bf16x16 bfm = frag_ld(sB[wv] + nt * 16 * 32, 32, lane);
        acc[mt][nt] = wmma32(af, bfm, acc[mt][nt]);
      }
    }
  }
  const float beta = betaP[0];
#pragma unroll
  for (int mt = 0; mt < 4; ++mt)
#pragma unroll
    for (int nt = 0; nt < 2; ++nt) {
      int u = nt * 16 + (lane & 15);
      if (u < NV) {
#pragma unroll
        for (int r = 0; r < 8; ++r) {
          int t = mt * 16 + 8 * (lane >> 4) + r;
          y[bc * TV + t * NV + u] = beta * acc[mt][nt][r];   // first writer
        }
      }
    }
}

// =====================================================================
// y2 (all subsets): per (b,t)   y[c,m] += gamma * sum_i sum_n x3_i[c,n]*x4_i[m,n]
// =====================================================================
__global__ __launch_bounds__(256) void y2_kernel(const unsigned short* __restrict__ x3b,
                                                 const unsigned short* __restrict__ x4b,
                                                 float* __restrict__ y,
                                                 const float* __restrict__ gammaP) {
  __shared__ __align__(16) unsigned short sA[256 * 32];
  __shared__ __align__(16) unsigned short sB[1024];
  const int bt = blockIdx.x;
  const int b = bt >> 6, t = bt & 63;
  const int tid = threadIdx.x, wv = tid >> 5, lane = tid & 31;
  const bool w0 = (__builtin_amdgcn_readfirstlane(wv) == 0);
  const long xbase = ((long)b * NC) * TV + t * NV;
  f32x8 acc[2][2];
#pragma unroll
  for (int mt = 0; mt < 2; ++mt)
#pragma unroll
    for (int nt = 0; nt < 2; ++nt) acc[mt][nt] = (f32x8){0, 0, 0, 0, 0, 0, 0, 0};
  for (int i = 0; i < NS; ++i) {
    if (w0) {
      tdm_load_2d(lds_off(&sA[0]), x3b + (long)i * NEX + xbase, 25, 256, 32, 256, TV);
      tdm_load_2d(lds_off(&sB[0]), x4b + (long)i * X4N + (long)bt * (NV * NV), 25, 25, 32, 32, 25);
      tdm_wait0();
    }
    __syncthreads();
#pragma unroll
    for (int mt = 0; mt < 2; ++mt) {
      int c0 = wv * 32 + mt * 16;
      bf16x16 af = frag_ld(sA + c0 * 32, 32, lane);
#pragma unroll
      for (int nt = 0; nt < 2; ++nt) {
        bf16x16 bfm = frag_ld(sB + nt * 16 * 32, 32, lane);
        acc[mt][nt] = wmma32(af, bfm, acc[mt][nt]);
      }
    }
    __syncthreads();
  }
  const float gamma = gammaP[0];
#pragma unroll
  for (int mt = 0; mt < 2; ++mt) {
    int c0 = wv * 32 + mt * 16;
#pragma unroll
    for (int nt = 0; nt < 2; ++nt) {
      int m = nt * 16 + (lane & 15);
      if (m < NV) {
#pragma unroll
        for (int r = 0; r < 8; ++r) {
          int c = c0 + 8 * (lane >> 4) + r;
          y[((long)(b * NC + c)) * TV + t * NV + m] += gamma * acc[mt][nt][r];
        }
      }
    }
  }
}

// =====================================================================
// BatchNorm over (B,T,V) per channel + residual + relu -> local
// =====================================================================
__global__ __launch_bounds__(256) void bn_local_kernel(const float* __restrict__ y,
                                                       const float* __restrict__ x,
                                                       const float* __restrict__ bw,
                                                       const float* __restrict__ bb,
                                                       float* __restrict__ localp) {
  __shared__ float r1[256], r2[256];
  const int c = blockIdx.x, tid = threadIdx.x;
  float s1 = 0.f, s2 = 0.f;
  for (int idx = tid; idx < NB * TV; idx += 256) {
    int b = idx / TV, tv = idx % TV;
    float v = y[((long)b * NC + c) * TV + tv];
    s1 += v; s2 += v * v;
  }
  r1[tid] = s1; r2[tid] = s2; __syncthreads();
  for (int off = 128; off; off >>= 1) {
    if (tid < off) { r1[tid] += r1[tid + off]; r2[tid] += r2[tid + off]; }
    __syncthreads();
  }
  const float inv = 1.f / (NB * TV);
  float mean = r1[0] * inv;
  float var = r2[0] * inv - mean * mean;
  float w = bw[c] * rsqrtf(var + 1e-5f);
  float bias = bb[c] - mean * w;
  for (int idx = tid; idx < NB * TV; idx += 256) {
    int b = idx / TV, tv = idx % TV;
    long o = ((long)b * NC + c) * TV + tv;
    localp[o] = fmaxf(y[o] * w + bias + x[o], 0.f);
  }
}

// ---- LayerNorm over C, wave per row -> hn bf16 ----
__global__ __launch_bounds__(256) void ln1_kernel(const unsigned short* __restrict__ xt,
                                                  const float* __restrict__ lw,
                                                  const float* __restrict__ lb,
                                                  unsigned short* __restrict__ hn) {
  const int wv = threadIdx.x >> 5, lane = threadIdx.x & 31;
  const long row = (long)blockIdx.x * 8 + wv;
  const unsigned short* src = xt + row * NC + lane * 8;
  float v[8]; float s = 0.f;
#pragma unroll
  for (int j = 0; j < 8; ++j) { v[j] = bf2f(src[j]); s += v[j]; }
  for (int off = 16; off; off >>= 1) s += __shfl_xor(s, off, 32);
  float mean = s * (1.f / NC);
  float q = 0.f;
#pragma unroll
  for (int j = 0; j < 8; ++j) { float d = v[j] - mean; q += d * d; }
  for (int off = 16; off; off >>= 1) q += __shfl_xor(q, off, 32);
  float rs = rsqrtf(q * (1.f / NC) + 1e-5f);
  unsigned short* dst = hn + row * NC + lane * 8;
#pragma unroll
  for (int j = 0; j < 8; ++j) {
    int ci = lane * 8 + j;
    dst[j] = f2bf((v[j] - mean) * rs * lw[ci] + lb[ci]);
  }
}

// =====================================================================
// Attention: per (b*t, head): TDM Q/K tiles -> QK^T (WMMA) -> +RPE bias,
// softmax -> attn f32 to d_out + bf16 to LDS -> attn@V (WMMA) -> o bf16
// =====================================================================
__global__ __launch_bounds__(128) void attn_kernel(const unsigned short* __restrict__ qb,
                                                   const unsigned short* __restrict__ kb,
                                                   const unsigned short* __restrict__ vb,
                                                   const int* __restrict__ hop,
                                                   const float* __restrict__ rpe,
                                                   float* __restrict__ attnOut,
                                                   unsigned short* __restrict__ ob) {
  __shared__ __align__(16) unsigned short sQV[4][2048];  // Q (32x64) then V^T (64x32)
  __shared__ __align__(16) unsigned short sK[4][2048];   // K (32x64)
  __shared__ __align__(16) unsigned short sAt[4][1024];  // attn bf16 (32x32)
  __shared__ __align__(16) float sS[4][1024];            // scores f32 (32x32)
  const int bt = blockIdx.x;
  const int h = threadIdx.x >> 5, lane = threadIdx.x & 31;
  const long rowb = (long)bt * NV;
  tdm_load_2d(lds_off(&sQV[h][0]), qb + rowb * NC + h * 64, 64, 25, 64, 32, NC);
  tdm_load_2d(lds_off(&sK[h][0]),  kb + rowb * NC + h * 64, 64, 25, 64, 32, NC);
  for (int idx = lane; idx < 1024; idx += 32) sAt[h][idx] = 0;
  tdm_wait0();
  __syncthreads();
  f32x8 sc[2][2];
#pragma unroll
  for (int mt = 0; mt < 2; ++mt)
#pragma unroll
    for (int nt = 0; nt < 2; ++nt) sc[mt][nt] = (f32x8){0, 0, 0, 0, 0, 0, 0, 0};
  for (int kk = 0; kk < 64; kk += 32) {
#pragma unroll
    for (int mt = 0; mt < 2; ++mt) {
      bf16x16 af = frag_ld(sQV[h] + mt * 16 * 64 + kk, 64, lane);
#pragma unroll
      for (int nt = 0; nt < 2; ++nt) {
        bf16x16 bfm = frag_ld(sK[h] + nt * 16 * 64 + kk, 64, lane);
        sc[mt][nt] = wmma32(af, bfm, sc[mt][nt]);
      }
    }
  }
  {
    int mb = 8 * (lane >> 4), nb = lane & 15;
#pragma unroll
    for (int mt = 0; mt < 2; ++mt)
#pragma unroll
      for (int nt = 0; nt < 2; ++nt)
#pragma unroll
        for (int r = 0; r < 8; ++r)
          sS[h][(mt * 16 + mb + r) * 32 + nt * 16 + nb] = sc[mt][nt][r];
  }
  __syncthreads();
  if (lane < NV) {
    const int i = lane;
    float p[NV]; float mx = -1e30f;
#pragma unroll
    for (int j = 0; j < NV; ++j) {
      float s = (sS[h][i * 32 + j] + rpe[h * NHOP + hop[i * NV + j]]) * 0.125f;
      p[j] = s; mx = fmaxf(mx, s);
    }
    float sum = 0.f;
#pragma unroll
    for (int j = 0; j < NV; ++j) { float e = expf(p[j] - mx); p[j] = e; sum += e; }
    float inv = 1.f / sum;
    float* aout = attnOut + ((long)bt * NH + h) * (NV * NV) + i * NV;
#pragma unroll
    for (int j = 0; j < NV; ++j) {
      float pr = p[j] * inv;
      aout[j] = pr;
      sAt[h][i * 32 + j] = f2bf(pr);
    }
  }
  __syncthreads();
  for (int idx = lane; idx < 2048; idx += 32) {  // V^T: rows d(64) x cols j(32)
    int d = idx >> 5, j = idx & 31;
    sQV[h][idx] = (j < NV) ? vb[(rowb + j) * NC + h * 64 + d] : (unsigned short)0;
  }
  __syncthreads();
#pragma unroll
  for (int mt = 0; mt < 2; ++mt) {
    bf16x16 af = frag_ld(sAt[h] + mt * 16 * 32, 32, lane);
#pragma unroll
    for (int nt = 0; nt < 4; ++nt) {
      bf16x16 bfm = frag_ld(sQV[h] + nt * 16 * 32, 32, lane);
      f32x8 acc = {0, 0, 0, 0, 0, 0, 0, 0};
      acc = wmma32(af, bfm, acc);
      int nb = nt * 16 + (lane & 15);
#pragma unroll
      for (int r = 0; r < 8; ++r) {
        int m = mt * 16 + 8 * (lane >> 4) + r;
        if (m < NV) ob[(rowb + m) * NC + h * 64 + nb] = f2bf(acc[r]);
      }
    }
  }
}

// ---- LayerNorm over GH=64, wave per row -> bf16 ----
__global__ __launch_bounds__(256) void gln_kernel(const float* __restrict__ hg,
                                                  const float* __restrict__ gw,
                                                  const float* __restrict__ gb,
                                                  unsigned short* __restrict__ out) {
  const int wv = threadIdx.x >> 5, lane = threadIdx.x & 31;
  const long row = (long)blockIdx.x * 8 + wv;
  float v0 = hg[row * NGH + lane];
  float v1 = hg[row * NGH + lane + 32];
  float s = v0 + v1;
  for (int off = 16; off; off >>= 1) s += __shfl_xor(s, off, 32);
  float mean = s * (1.f / NGH);
  float d0 = v0 - mean, d1 = v1 - mean;
  float q = d0 * d0 + d1 * d1;
  for (int off = 16; off; off >>= 1) q += __shfl_xor(q, off, 32);
  float rs = rsqrtf(q * (1.f / NGH) + 1e-5f);
  out[row * NGH + lane]      = f2bf(d0 * rs * gw[lane] + gb[lane]);
  out[row * NGH + lane + 32] = f2bf(d1 * rs * gw[lane + 32] + gb[lane + 32]);
}

// =====================================================================
// Launcher
// =====================================================================
extern "C" void kernel_launch(void* const* d_in, const int* in_sizes, int n_in,
                              void* d_out, int out_size, void* d_ws, size_t ws_size,
                              hipStream_t stream) {
  (void)in_sizes; (void)n_in; (void)out_size; (void)ws_size;
  const float* x     = (const float*)d_in[0];
  const int*   hop   = (const int*)d_in[1];
  const float* PA    = (const float*)d_in[2];
  const float* alpha = (const float*)d_in[3];
  const float* beta  = (const float*)d_in[4];
  const float* gamma = (const float*)d_in[5];
  const float* c1w   = (const float*)d_in[6];
  const float* c1b   = (const float*)d_in[7];
  const float* c3w   = (const float*)d_in[8];
  const float* c3b   = (const float*)d_in[9];
  const float* c4w   = (const float*)d_in[10];
  const float* c4b   = (const float*)d_in[11];
  const float* bn_w  = (const float*)d_in[12];
  const float* bn_b  = (const float*)d_in[13];
  const float* ln1_w = (const float*)d_in[14];
  const float* ln1_b = (const float*)d_in[15];
  const float* qkv_w = (const float*)d_in[16];
  const float* out_w = (const float*)d_in[17];
  const float* out_b = (const float*)d_in[18];
  const float* rpe   = (const float*)d_in[19];
  const float* g1_w  = (const float*)d_in[20];
  const float* g1_b  = (const float*)d_in[21];
  const float* gln_w = (const float*)d_in[22];
  const float* gln_b = (const float*)d_in[23];
  const float* g2_w  = (const float*)d_in[24];
  const float* g2_b  = (const float*)d_in[25];

  char* ws = (char*)d_ws;
  // ---- fixed region ----
  unsigned short* xtb    = (unsigned short*)(ws + 0);            // 52,428,800
  float*          xmean  = (float*)(ws + 52428800);              //  1,638,400
  float*          w3m    = (float*)(ws + 54067200);              //      3,328
  float*          b3m    = w3m + NS * NC;
  float*          x1m    = (float*)(ws + 54070528);              //    204,800
  float*          cm     = (float*)(ws + 54275328);              //    409,600
  unsigned short* c3wb   = (unsigned short*)(ws + 54684928);     //    393,216
  unsigned short* qkvwb  = (unsigned short*)(ws + 55078144);     //    393,216
  unsigned short* outwb  = (unsigned short*)(ws + 55471360);     //    131,072
  unsigned short* g1wb   = (unsigned short*)(ws + 55602432);     //     65,536
  unsigned short* g2wb   = (unsigned short*)(ws + 55667968);     //     32,768
  float*          localp = (float*)(ws + 55700736);              // 104,857,600
  float*          gbuf   = (float*)(ws + 160558336);             // 104,857,600
  char*           S      = ws + 265415936;                       // 338,944,000 reused region
  // phase L (all 3 subsets resident)
  float*          ybuf   = (float*)(S + 0);                      // 104,857,600
  unsigned short* x3bf   = (unsigned short*)(S + 104857600);     // 3 x 52,428,800
  unsigned short* abf    = (unsigned short*)(S + 262144000);     // 3 x 20,480,000
  unsigned short* x4bf   = (unsigned short*)(S + 323584000);     // 3 x  5,120,000
  // phase G (after bn_local, phase-L buffers dead)
  unsigned short* hnbf   = (unsigned short*)(S + 0);
  unsigned short* qbf    = (unsigned short*)(S + 52428800);
  unsigned short* kbf    = (unsigned short*)(S + 104857600);
  unsigned short* vbf    = (unsigned short*)(S + 157286400);
  unsigned short* obf    = (unsigned short*)(S + 0);             // reuse hnbf slot
  float*          hg     = (float*)(S + 157286400);              // reuse v slot
  unsigned short* hglnbf = (unsigned short*)(S + 183500800);

  float* fused   = (float*)d_out;
  float* attnOut = (float*)d_out + NEX;

  // ---- weight conversions ----
  cvt_f32_bf16<<<(NS * NC * NC + 255) / 256, 256, 0, stream>>>(c3w, c3wb, (long)NS * NC * NC);
  cvt_f32_bf16<<<(3 * NC * NC + 255) / 256, 256, 0, stream>>>(qkv_w, qkvwb, (long)3 * NC * NC);
  cvt_f32_bf16<<<(NC * NC + 255) / 256, 256, 0, stream>>>(out_w, outwb, (long)NC * NC);
  cvt_f32_bf16<<<(NGH * 2 * NC + 255) / 256, 256, 0, stream>>>(g1_w, g1wb, (long)NGH * 2 * NC);
  cvt_f32_bf16<<<(NC * NGH + 255) / 256, 256, 0, stream>>>(g2_w, g2wb, (long)NC * NGH);

  // ---- prep ----
  xt_kernel<<<(int)(NEX / 256), 256, 0, stream>>>(x, xtb);
  xmean_kernel<<<NB * NC * NV / 256, 256, 0, stream>>>(x, xmean);
  w3m_kernel<<<(NS * (NC + 1) + 255) / 256, 256, 0, stream>>>(c3w, c3b, w3m, b3m);

  // ---- local stream: build all 3 subsets, then single-pass y1/y2 ----
  for (int i = 0; i < NS; ++i) {
    EpiX3 e3{c3b + i * NC, x3bf + (long)i * NEX};
    gemm_nt<EpiX3><<<dim3(TV / 64, NC / 64, NB), 256, 0, stream>>>(
        xtb, c3wb + (long)i * NC * NC, NC, (long)TV * NC, e3);
    x1m_kernel<<<NB * NR * NV / 256, 256, 0, stream>>>(c1w, c1b, xmean, x1m, i);
    a_kernel<<<NB * NV * NV, 256, 0, stream>>>(c4w, c4b, x1m, PA, alpha, abf + (long)i * ABFN, i);
    cm_kernel<<<NB * TV / 256, 256, 0, stream>>>(x, w3m, b3m, cm, i);
    x4_kernel<<<(int)(X4N / 256), 256, 0, stream>>>(cm, x4bf + (long)i * X4N);
  }
  y1_kernel<<<NB * NC / 8, 256, 0, stream>>>(x3bf, abf, ybuf, beta);
  y2_kernel<<<NB * NT, 256, 0, stream>>>(x3bf, x4bf, ybuf, gamma);
  bn_local_kernel<<<NC, 256, 0, stream>>>(ybuf, x, bn_w, bn_b, localp);

  // ---- global stream ----
  ln1_kernel<<<MR / 8, 256, 0, stream>>>(xtb, ln1_w, ln1_b, hnbf);
  EpiQKV eq{qbf, kbf, vbf};
  gemm_nt<EpiQKV><<<dim3(MR / 64, 768 / 64, 1), 256, 0, stream>>>(hnbf, qkvwb, NC, 0L, eq);
  attn_kernel<<<NB * NT, 128, 0, stream>>>(qbf, kbf, vbf, hop, rpe, attnOut, obf);
  EpiOutProj eo{out_b, x, gbuf};
  gemm_nt<EpiOutProj><<<dim3(MR / 64, NC / 64, 1), 256, 0, stream>>>(obf, outwb, NC, 0L, eo);

  // ---- gated fusion (concat fused into g1 GEMM A-loader) ----
  gemm_g1<<<MR / 64, 256, 0, stream>>>(localp, gbuf, g1wb, g1_b, hg);
  gln_kernel<<<MR / 8, 256, 0, stream>>>(hg, gln_w, gln_b, hglnbf);
  EpiG2Fuse e2{g2_b, localp, gbuf, fused};
  gemm_nt<EpiG2Fuse><<<dim3(MR / 64, NC / 64, 1), 256, 0, stream>>>(hglnbf, g2wb, NGH, 0L, e2);
}
